// CRF_58557584114171
// MI455X (gfx1250) — matrix operand order repaired and checked by reading, not compiled, not run on hardware
//
#include <hip/hip_runtime.h>

// CRF log-likelihood for S=512, B=128, T=256 on gfx1250 (MI455X).
// Forward algorithm recast as exp-space GEMM per step, run on v_wmma_f32_16x16x32_f16.

typedef __attribute__((ext_vector_type(16))) _Float16 v16h;
typedef __attribute__((ext_vector_type(8)))  float    v8f;

#define S_LEN 512
#define BB    128
#define T     256
#define ROWS  16          // batch rows per workgroup
#define SC    260         // C32 row stride (floats), padded vs 256 for bank spread
#define SA    264         // A16 row stride (halves), padded vs 256 for bank spread

union U32H2 { unsigned u; _Float16 h[2]; };

__device__ __forceinline__ float wave_max(float v) {
#pragma unroll
  for (int o = 16; o >= 1; o >>= 1) v = fmaxf(v, __shfl_xor(v, o, 32));
  return v;
}
__device__ __forceinline__ float wave_sum(float v) {
#pragma unroll
  for (int o = 16; o >= 1; o >>= 1) v += __shfl_xor(v, o, 32);
  return v;
}

// ---------------------------------------------------------------------------
// Kernel 0: GT[j*T + k] = (f16) exp(transitions[k*T + j])   (transposed so the
// WMMA B-fragment gather reads contiguous K pairs).
// ---------------------------------------------------------------------------
__global__ void crf_prep(const float* __restrict__ trans, _Float16* __restrict__ GT) {
  int idx = blockIdx.x * blockDim.x + threadIdx.x;   // 65536 threads
  int j = idx >> 8, k = idx & 255;
  GT[j * T + k] = (_Float16)__expf(trans[k * T + j]);
}

// ---------------------------------------------------------------------------
// Kernel 1: numerator score per batch element (gather path score).
// ---------------------------------------------------------------------------
__global__ void crf_numer(const float* __restrict__ em, const int* __restrict__ tags,
                          const int* __restrict__ amask, const float* __restrict__ startT,
                          const float* __restrict__ endT, const float* __restrict__ trans,
                          float* __restrict__ num) {
  int b = threadIdx.x;
  if (b >= BB) return;
  int t0 = tags[b];
  float score = startT[t0] + em[(size_t)b * T + t0];
  int prev = amask[b] ? t0 : 0;
  int cnt  = (amask[b] != 0);
  for (int s = 1; s < S_LEN; ++s) {
    int mk  = amask[s * BB + b];
    int cur = mk ? tags[s * BB + b] : 0;
    if (mk) {
      score += trans[prev * T + cur] + em[((size_t)s * BB + b) * T + cur];
      ++cnt;
    }
    prev = cur;
  }
  int last = tags[(cnt - 1) * BB + b];
  num[b] = score + endT[last];
}

// ---------------------------------------------------------------------------
// Kernel 2: forward algorithm. 8 workgroups x 512 threads (16 waves).
// Workgroup owns 16 batch rows; wave w owns output columns [16w,16w+16).
// Per step: rowmax -> A16=exp(carry-rowmax) -> 8x WMMA f16 -> em + log fixup.
// ---------------------------------------------------------------------------
__global__ __launch_bounds__(512) void crf_forward(
    const float* __restrict__ em, const int* __restrict__ amask,
    const float* __restrict__ startT, const float* __restrict__ endT,
    const _Float16* __restrict__ GT, float* __restrict__ den) {
  __shared__ float    C32[ROWS * SC];
  __shared__ _Float16 A16[ROWS * SA];
  __shared__ float    rmax[ROWS];

  const int tid  = threadIdx.x;
  const int wave = tid >> 5;
  const int lane = tid & 31;
  const int lhi  = (lane < 16) ? 0 : 1;   // high half-wave flag
  const int lm   = lane & 15;
  const int rb   = blockIdx.x * ROWS;     // first batch row of this WG
  const int jb   = wave * 16;             // first output column of this wave

  // ---- Preload B fragments (exp(trans)^T columns jb..jb+15) into registers.
  // B 32x16 f16 k order here (contiguous, as A): VGPR v halves h: K = (lane<16?0:16) + 2v + h, N = lane%16.
  v16h Bf[8];
#pragma unroll
  for (int kt = 0; kt < 8; ++kt) {
#pragma unroll
    for (int v = 0; v < 8; ++v) {
      int k = 32 * kt + (lhi ? 16 : 0) + 2 * v;
      U32H2 u;
      u.u = *(const unsigned*)(GT + ((size_t)(jb + lm) * T + k));
      Bf[kt][2 * v]     = u.h[0];
      Bf[kt][2 * v + 1] = u.h[1];
    }
  }

  // ---- carry init: start_transitions + emissions[0]
  for (int i = tid; i < ROWS * T; i += 512) {
    int r = i >> 8, t = i & 255;
    C32[r * SC + t] = startT[t] + em[(size_t)(rb + r) * T + t];
  }
  __syncthreads();

  for (int s = 1; s < S_LEN; ++s) {
    // -- rowmax + A16 for row r == wave id
    {
      const int r = wave;
      float vals[8], mv = -INFINITY;
#pragma unroll
      for (int i = 0; i < 8; ++i) {
        vals[i] = C32[r * SC + lane + 32 * i];
        mv = fmaxf(mv, vals[i]);
      }
      mv = wave_max(mv);
#pragma unroll
      for (int i = 0; i < 8; ++i)
        A16[r * SA + lane + 32 * i] = (_Float16)__expf(vals[i] - mv);
      if (lane == 0) rmax[r] = mv;
    }
    __syncthreads();

    // -- prefetch next step's emissions tile for this wave (global_prefetch_b8)
    if (s + 1 < S_LEN)
      __builtin_prefetch(&em[((size_t)(s + 1) * BB + rb + lm) * T + jb], 0, 0);

    // -- 16x16 tile GEMM over K=256: 8x v_wmma_f32_16x16x32_f16
    v8f acc = {};
#pragma unroll
    for (int kt = 0; kt < 8; ++kt) {
      // A 16x32 f16 k order here (contiguous, as B): M = lane%16; VGPR v halves h:
      //   K = 32*kt + 2v + h + (lane<16 ? 0 : 16)
      v16h Af;
#pragma unroll
      for (int v = 0; v < 8; ++v) {
        int kb = 32 * kt + 2 * v + (lhi ? 16 : 0);
        U32H2 u;
        u.u = *(const unsigned*)&A16[lm * SA + kb];
        Af[2 * v]     = u.h[0];
        Af[2 * v + 1] = u.h[1];
      }
      acc = __builtin_amdgcn_wmma_f32_16x16x32_f16(
          false, Af, false, Bf[kt], (short)0, acc, false, false);
    }

    // -- write back: carry[r][j] = mask ? em + rowmax + log(P) : carry
    // D layout: VGPR o -> M = o + (lane<16?0:8); N = lane%16.
    const size_t sbase = (size_t)s * BB;
#pragma unroll
    for (int o = 0; o < 8; ++o) {
      int r = o + (lhi ? 8 : 0);
      int b = rb + r;
      int j = jb + lm;
      int mk = amask[sbase + b];
      float nv;
      if (mk) {
        float e = em[(sbase + b) * T + j];
        nv = e + rmax[r] + __logf(acc[o]);
      } else {
        nv = C32[r * SC + j];
      }
      C32[r * SC + j] = nv;
    }
    __syncthreads();
  }

  // ---- denominator: logsumexp_j(carry[r][j] + end_transitions[j]), row r = wave
  {
    const int r = wave;
    float vals[8], mv = -INFINITY;
#pragma unroll
    for (int i = 0; i < 8; ++i) {
      vals[i] = C32[r * SC + lane + 32 * i] + endT[lane + 32 * i];
      mv = fmaxf(mv, vals[i]);
    }
    mv = wave_max(mv);
    float sum = 0.f;
#pragma unroll
    for (int i = 0; i < 8; ++i) sum += __expf(vals[i] - mv);
    sum = wave_sum(sum);
    if (lane == 0) den[rb + r] = mv + __logf(sum);
  }
}

// ---------------------------------------------------------------------------
// Kernel 3: deterministic final reduction  out = sum_b (num[b] - den[b])
// ---------------------------------------------------------------------------
__global__ void crf_final(const float* __restrict__ num, const float* __restrict__ den,
                          float* __restrict__ out) {
  __shared__ float red[BB];
  int b = threadIdx.x;
  red[b] = num[b] - den[b];
  __syncthreads();
#pragma unroll
  for (int st = 64; st >= 1; st >>= 1) {
    if (b < st) red[b] += red[b + st];
    __syncthreads();
  }
  if (b == 0) out[0] = red[0];
}

// ---------------------------------------------------------------------------
extern "C" void kernel_launch(void* const* d_in, const int* in_sizes, int n_in,
                              void* d_out, int out_size, void* d_ws, size_t ws_size,
                              hipStream_t stream) {
  const float* em     = (const float*)d_in[0];   // (S,B,T) f32
  const int*   tags   = (const int*)d_in[1];     // (S,B)   i32
  const int*   amask  = (const int*)d_in[2];     // (S,B)   i32
  const float* startT = (const float*)d_in[3];   // (T,)
  const float* endT   = (const float*)d_in[4];   // (T,)
  const float* trans  = (const float*)d_in[5];   // (T,T)
  float* out = (float*)d_out;

  // workspace layout: [GT f16: T*T*2 = 131072 B][num: 128 f32][den: 128 f32]
  _Float16* GT  = (_Float16*)d_ws;
  float*    num = (float*)((char*)d_ws + (size_t)T * T * sizeof(_Float16));
  float*    den = num + BB;

  crf_prep  <<<T * T / 256, 256, 0, stream>>>(trans, GT);
  crf_numer <<<1, BB, 0, stream>>>(em, tags, amask, startT, endT, trans, num);
  crf_forward<<<BB / ROWS, 512, 0, stream>>>(em, amask, startT, endT, GT, den);
  crf_final <<<1, BB, 0, stream>>>(num, den, out);
}
